// FullAttentionHead_18528488915523
// MI455X (gfx1250) — compile-verified, matrix operand order checked by reading, not run on hardware
//
#include <hip/hip_runtime.h>
#include <hip/hip_bf16.h>
#include <math.h>
#include <stdint.h>

// Problem constants (match reference)
static constexpr int Bn = 4;
static constexpr int Nn = 4096;
static constexpr int En = 1024;
static constexpr int Hn = 64;

typedef __attribute__((ext_vector_type(16))) _Float16 v16h;
typedef __attribute__((ext_vector_type(8)))  float    v8f;

__device__ __forceinline__ v8f wmma_f16(v16h a, v16h b, v8f c) {
    // D = A(16x32 f16) * B(32x16 f16) + C(16x16 f32)
    return __builtin_amdgcn_wmma_f32_16x16x32_f16(
        /*neg_a=*/false, a, /*neg_b=*/false, b,
        /*c_mod=*/(short)0, c, /*reuse_a=*/false, /*reuse_b=*/false);
}

// K-index of the even element of fragment register-pair r for lane-group g
// (16-bit A/B fragment layout, CDNA5 ISA §7.12.2)
__device__ __forceinline__ int frag_k(int r, int g) {
    return ((r >> 2) << 4) + ((r & 3) << 1) + (g << 3);
}

__device__ __forceinline__ v8f vzero8() {
    v8f z = {0.f, 0.f, 0.f, 0.f, 0.f, 0.f, 0.f, 0.f};
    return z;
}

// Async DMA: 16 bytes per lane, global -> LDS. Tracked by ASYNCcnt.
// (CDNA5 ISA opcode 98 GLOBAL_LOAD_ASYNC_TO_LDS_B128; generic LDS
//  pointers carry the LDS byte offset in their low 32 bits, ISA §10.2.)
__device__ __forceinline__ void async_copy_b128(uint32_t lds_off, const void* gaddr) {
    asm volatile("global_load_async_to_lds_b128 %0, %1, off"
                 :: "v"(lds_off), "v"((uint64_t)(uintptr_t)gaddr)
                 : "memory");
}

// --------------------------------------------------------------------------
// Kernel 0: convert Wq|Wk|Wv (f32, each 64x1024) to one packed f16 buffer.
// --------------------------------------------------------------------------
__global__ void __launch_bounds__(256)
wconv_kernel(const float* __restrict__ Wq, const float* __restrict__ Wk,
             const float* __restrict__ Wv, _Float16* __restrict__ wH) {
    const int i = blockIdx.x * 256 + threadIdx.x;     // 0 .. 3*64*1024-1
    const int per = Hn * En;
    if (i < 3 * per) {
        const float* src = (i < per) ? Wq : ((i < 2 * per) ? Wk : Wv);
        wH[i] = (_Float16)src[i % per];
    }
}

// --------------------------------------------------------------------------
// Kernel 1: q = x Wq^T, k = x Wk^T, v = x Wv^T  (f32 in, f16 out)
// 4 waves/block, one 16-token tile each. Per-K-chunk weight slice is DMA'd
// into LDS (double-buffered, ASYNCcnt) and shared by all 4 waves.
// V is stored transposed (B, H, N) so PV B-fragments are contiguous later.
// --------------------------------------------------------------------------
__global__ void __launch_bounds__(128)
qkv_proj_kernel(const float* __restrict__ x,
                const _Float16* __restrict__ wH,   // [3][64][1024] f16
                _Float16* __restrict__ qh,         // (B, N, H)
                _Float16* __restrict__ kh,         // (B, N, H)
                _Float16* __restrict__ vT)         // (B, H, N)
{
    __shared__ __align__(16) _Float16 wbuf[2][3][Hn][32];   // 2 x 12 KB

    const int tid  = threadIdx.x;
    const int wave = tid >> 5;
    const int lane = tid & 31;
    const int tile = blockIdx.x * 4 + wave;     // 0 .. B*N/16 - 1
    const int rowBase = tile * 16;
    const int b    = rowBase / Nn;
    const int tok0 = rowBase % Nn;
    const int m = lane & 15;
    const int g = lane >> 4;

    // Stage the (3 x 64 heads x 32 k) f16 weight slice for chunk kb into
    // LDS buffer s: 768 16-byte segments, 6 per thread -> 6 async ops/wave.
    auto stage = [&](int kb, int s) {
#pragma unroll
        for (int it = 0; it < 6; ++it) {
            const int flat = it * 128 + tid;    // 0..767
            const int mu   = flat >> 8;         // which of Wq/Wk/Wv
            const int rem  = flat & 255;
            const int row  = rem >> 2;          // head row 0..63
            const int seg  = rem & 3;           // 8-half segment within chunk
            const _Float16* src = wH + ((size_t)(mu * Hn + row)) * En + kb + seg * 8;
            const uint32_t  dst = (uint32_t)(uintptr_t)(&wbuf[s][mu][row][seg * 8]);
            async_copy_b128(dst, src);
        }
    };

    v8f accQ[4], accK[4], accV[4];
#pragma unroll
    for (int t = 0; t < 4; ++t) { accQ[t] = vzero8(); accK[t] = vzero8(); accV[t] = vzero8(); }

    const float* xrow = x + ((size_t)b * Nn + tok0 + m) * En;

    stage(0, 0);
    for (int kb = 0; kb < En; kb += 32) {
        const int cur = (kb >> 5) & 1;
        if (kb + 32 < En) {
            stage(kb + 32, cur ^ 1);
            asm volatile("s_wait_asynccnt 0x6" ::: "memory");  // cur chunk landed
        } else {
            asm volatile("s_wait_asynccnt 0x0" ::: "memory");
        }
        __syncthreads();

        // A fragment: 16x32 tile of x rows, f32 -> f16 (pairs -> b128 loads)
        v16h a;
#pragma unroll
        for (int r = 0; r < 8; ++r) {
            const int k0 = kb + frag_k(r, g);
            float2 t2 = *(const float2*)(xrow + k0);
            a[2 * r]     = (_Float16)t2.x;
            a[2 * r + 1] = (_Float16)t2.y;
        }
        if (kb + 32 < En) __builtin_prefetch(xrow + kb + 32, 0, 3);

#pragma unroll
        for (int t = 0; t < 4; ++t) {
            const int head = t * 16 + m;        // B-fragment: n = lane%16
            v16h bq, bk, bv;
#pragma unroll
            for (int r = 0; r < 8; ++r) {
                const int k0 = frag_k(r, g);
                bq[2 * r] = wbuf[cur][0][head][k0]; bq[2 * r + 1] = wbuf[cur][0][head][k0 + 1];
                bk[2 * r] = wbuf[cur][1][head][k0]; bk[2 * r + 1] = wbuf[cur][1][head][k0 + 1];
                bv[2 * r] = wbuf[cur][2][head][k0]; bv[2 * r + 1] = wbuf[cur][2][head][k0 + 1];
            }
            accQ[t] = wmma_f16(a, bq, accQ[t]);
            accK[t] = wmma_f16(a, bk, accK[t]);
            accV[t] = wmma_f16(a, bv, accV[t]);
        }
        __syncthreads();   // all waves done reading wbuf[cur] before overwrite
    }

    // Store C tiles: lane holds row (r + 8g), column (lane%16) of each 16x16 tile
#pragma unroll
    for (int t = 0; t < 4; ++t) {
#pragma unroll
        for (int r = 0; r < 8; ++r) {
            const int row    = r + 8 * g;
            const size_t tok = (size_t)tok0 + row;
            const int head   = t * 16 + m;
            qh[((size_t)b * Nn + tok) * Hn + head] = (_Float16)accQ[t][r];
            kh[((size_t)b * Nn + tok) * Hn + head] = (_Float16)accK[t][r];
            vT[((size_t)b * Hn + head) * Nn + tok] = (_Float16)accV[t][r];
        }
    }
}

// --------------------------------------------------------------------------
// Kernel 2: flash-attention. One wave per 16-query tile, 32 keys / iteration.
// S = (q k^T) * 1/sqrt(H), online softmax, O += P v.  P goes through a
// per-wave LDS tile (wave-local s_wait_dscnt, no block barrier) to convert
// the f32 C-layout into the f16 A-fragment layout.
// --------------------------------------------------------------------------
__global__ void __launch_bounds__(128)
flash_attn_kernel(const _Float16* __restrict__ qh,   // (B, N, H)
                  const _Float16* __restrict__ kh,   // (B, N, H)
                  const _Float16* __restrict__ vT,   // (B, H, N)
                  const int* __restrict__ mask,      // (B, N)
                  float* __restrict__ out)           // (B, N, H)
{
    __shared__ __align__(16) _Float16 Pld[4][16][36]; // per-wave 16x32 P (+pad)

    const int wave = threadIdx.x >> 5;
    const int lane = threadIdx.x & 31;
    const int qt = blockIdx.x * 4 + wave;             // 0 .. B*N/16 - 1
    const int b  = qt >> 8;                           // N/16 = 256 q-tiles/batch
    const int q0 = (qt & 255) * 16;
    const int m = lane & 15;
    const int g = lane >> 4;
    const float scale = 0.125f;                       // 1/sqrt(64)

    v8f o[4];
#pragma unroll
    for (int t = 0; t < 4; ++t) o[t] = vzero8();
    float mRow[8], lRow[8];
#pragma unroll
    for (int r = 0; r < 8; ++r) { mRow[r] = -INFINITY; lRow[r] = 0.f; }

    // Preload the query A-fragments (two K-steps over H=64)
    v16h aq[2];
    const _Float16* qrow = qh + ((size_t)b * Nn + q0 + m) * Hn;
#pragma unroll
    for (int ks = 0; ks < 2; ++ks) {
#pragma unroll
        for (int r = 0; r < 8; ++r) {
            const int k0 = ks * 32 + frag_k(r, g);
            aq[ks][2 * r]     = qrow[k0];
            aq[ks][2 * r + 1] = qrow[k0 + 1];
        }
    }

    for (int key0 = 0; key0 < Nn; key0 += 32) {
        // Prefetch next key/value block while this one computes
        if (key0 + 32 < Nn) {
            __builtin_prefetch(kh + ((size_t)b * Nn + key0 + 32 + lane) * Hn, 0, 3);
            __builtin_prefetch(vT + ((size_t)b * Hn + lane) * Nn + key0 + 32, 0, 3);
            __builtin_prefetch(vT + ((size_t)b * Hn + 32 + lane) * Nn + key0 + 32, 0, 3);
        }

        // ---- S = q k^T (two 16x16 subtiles over 32 keys) ----
        v8f s[2];
        int mk[2];
#pragma unroll
        for (int j = 0; j < 2; ++j) {
            s[j] = vzero8();
#pragma unroll
            for (int ks = 0; ks < 2; ++ks) {
                v16h bk;
                const _Float16* krow =
                    kh + ((size_t)b * Nn + key0 + j * 16 + m) * Hn + ks * 32;
#pragma unroll
                for (int r = 0; r < 8; ++r) {
                    const int k0 = frag_k(r, g);
                    bk[2 * r]     = krow[k0];
                    bk[2 * r + 1] = krow[k0 + 1];
                }
                s[j] = wmma_f16(aq[ks], bk, s[j]);
            }
            mk[j] = mask[(size_t)b * Nn + key0 + j * 16 + m];
#pragma unroll
            for (int r = 0; r < 8; ++r) {
                float sv = s[j][r] * scale;
                s[j][r] = (mk[j] == 0) ? -INFINITY : sv;
            }
        }

        // ---- online softmax: row max across the 16 lanes of this lane-group ----
        float rm[8];
#pragma unroll
        for (int r = 0; r < 8; ++r) rm[r] = fmaxf(s[0][r], s[1][r]);
#pragma unroll
        for (int off = 1; off < 16; off <<= 1) {
#pragma unroll
            for (int r = 0; r < 8; ++r)
                rm[r] = fmaxf(rm[r], __shfl_xor(rm[r], off, 32));
        }
        float alpha[8];
#pragma unroll
        for (int r = 0; r < 8; ++r) {
            const float mNew = fmaxf(mRow[r], rm[r]);
            alpha[r] = (mNew == -INFINITY) ? 1.f : __expf(mRow[r] - mNew);
            mRow[r] = mNew;
        }

        // ---- P = exp(S - m), write to per-wave LDS tile ----
        float ps[8];
#pragma unroll
        for (int r = 0; r < 8; ++r) {
            const float p0 = (mRow[r] == -INFINITY) ? 0.f : __expf(s[0][r] - mRow[r]);
            const float p1 = (mRow[r] == -INFINITY) ? 0.f : __expf(s[1][r] - mRow[r]);
            Pld[wave][r + 8 * g][0 * 16 + m] = (_Float16)p0;
            Pld[wave][r + 8 * g][1 * 16 + m] = (_Float16)p1;
            ps[r] = p0 + p1;
        }
#pragma unroll
        for (int off = 1; off < 16; off <<= 1) {
#pragma unroll
            for (int r = 0; r < 8; ++r)
                ps[r] += __shfl_xor(ps[r], off, 32);
        }
#pragma unroll
        for (int r = 0; r < 8; ++r) lRow[r] = lRow[r] * alpha[r] + ps[r];

        // Wave-local LDS fence: stores visible to this wave's lanes (per-wave
        // tile, LDS ops are in-order per wave -> no block barrier needed).
        __builtin_amdgcn_wave_barrier();
        asm volatile("s_wait_dscnt 0x0" ::: "memory");

        // ---- reload P as an A fragment (16x32 f16) ----
        v16h ap;
#pragma unroll
        for (int r = 0; r < 8; ++r) {
            const int k0 = frag_k(r, g);
            ap[2 * r]     = Pld[wave][m][k0];
            ap[2 * r + 1] = Pld[wave][m][k0 + 1];
        }

        // ---- O = O*alpha + P V ----
#pragma unroll
        for (int t = 0; t < 4; ++t) {
            v16h bv;
            const _Float16* vrow =
                vT + ((size_t)b * Hn + t * 16 + m) * Nn + key0;
#pragma unroll
            for (int r = 0; r < 8; ++r) {
                const int k0 = frag_k(r, g);
                bv[2 * r]     = vrow[k0];
                bv[2 * r + 1] = vrow[k0 + 1];
            }
#pragma unroll
            for (int r = 0; r < 8; ++r) o[t][r] *= alpha[r];
            o[t] = wmma_f16(ap, bv, o[t]);
        }
        __builtin_amdgcn_wave_barrier();   // keep next-iter P stores behind ap loads
    }

    // ---- normalize and store (f32 output, (B, N, H)) ----
#pragma unroll
    for (int t = 0; t < 4; ++t) {
#pragma unroll
        for (int r = 0; r < 8; ++r) {
            const float denom = lRow[r];
            const float val = (denom > 0.f) ? (o[t][r] / denom) : 0.f;
            const int row = r + 8 * g;
            out[((size_t)b * Nn + q0 + row) * Hn + t * 16 + m] = val;
        }
    }
}

// --------------------------------------------------------------------------
extern "C" void kernel_launch(void* const* d_in, const int* in_sizes, int n_in,
                              void* d_out, int out_size, void* d_ws, size_t ws_size,
                              hipStream_t stream) {
    const float* x  = (const float*)d_in[0];
    const float* Wq = (const float*)d_in[1];
    const float* Wk = (const float*)d_in[2];
    const float* Wv = (const float*)d_in[3];
    const int* mask = (const int*)d_in[4];
    float* out = (float*)d_out;

    const size_t qkvElems = (size_t)Bn * Nn * Hn;        // 1M f16 elems each
    _Float16* qh = (_Float16*)d_ws;
    _Float16* kh = qh + qkvElems;
    _Float16* vT = kh + qkvElems;
    _Float16* wH = vT + qkvElems;                        // 3*64*1024 f16

    // Kernel 0: weights f32 -> f16 (3*64*1024 = 196608 elements)
    wconv_kernel<<<768, 256, 0, stream>>>(Wq, Wk, Wv, wH);
    // Kernel 1: B*N/16 = 1024 wave-tiles, 4 waves (128 threads) per block
    qkv_proj_kernel<<<256, 128, 0, stream>>>(x, wH, qh, kh, vT);
    // Kernel 2: flash attention
    flash_attn_kernel<<<256, 128, 0, stream>>>(qh, kh, vT, mask, out);
}
